// Net_11768210391048
// MI455X (gfx1250) — matrix operand order repaired and checked by reading, not compiled, not run on hardware
//
#include <hip/hip_runtime.h>
#include <hip/hip_bf16.h>
#include <math.h>

// ---------------------------------------------------------------------------
// MI455X (gfx1250, wave32) implementation.
// All matmuls use V_WMMA_F32_16X16X32_BF16 (bf16 A/B, fp32 accumulate).
// LDS tiles are stored in WMMA-fragment-swizzled layout so each operand
// fragment is a single contiguous 32B v16bf load (ds_load_b128 x2).
// Jacobian Frobenius term reduced via tr(U~ G U~^T), G = W1 W1^T (Gram),
// with W2 / G kept in global memory pre-swizzled to B-fragment layout.
// ---------------------------------------------------------------------------

typedef __attribute__((ext_vector_type(16))) __bf16 v16bf;
typedef __attribute__((ext_vector_type(4)))  __bf16 v4bf;
typedef __attribute__((ext_vector_type(8)))  float  v8f;

#define DEV static __device__ __forceinline__

// ---- sizes -----------------------------------------------------------------
constexpr int BATCH = 2048;
constexpr int NINP  = 30;
constexpr int NX    = 32;      // N_INP + H
constexpr int SZ    = 1120;    // N*(N+1) + 2N
constexpr int LL    = 30;      // L
constexpr int NTRI  = 528;     // 32*33/2

// ---- workspace layout (floats, all offsets 32B-aligned) --------------------
constexpr size_t OFF_RAW  = 0;
constexpr size_t OFF_XN   = OFF_RAW + (size_t)BATCH*SZ;
constexpr size_t OFF_H1   = OFF_XN  + (size_t)BATCH*SZ;
constexpr size_t OFF_S1   = OFF_H1  + (size_t)BATCH*128;
constexpr size_t OFF_H2   = OFF_S1  + (size_t)BATCH*128;
constexpr size_t OFF_S2   = OFF_H2  + (size_t)BATCH*128;
constexpr size_t OFF_Z    = OFF_S2  + (size_t)BATCH*128;   // ld = 32, cols 0..29 valid
constexpr size_t OFF_D1   = OFF_Z   + (size_t)BATCH*32;
constexpr size_t OFF_D2   = OFF_D1  + (size_t)BATCH*128;
constexpr size_t OFF_G    = OFF_D2  + (size_t)BATCH*128;   // 128x128 Gram W1 W1^T
constexpr size_t OFF_SCAL = OFF_G   + 128*128;             // ssc row sums (SZ x 2)
constexpr size_t OFF_SCD  = OFF_SCAL + 2*SZ;               // scaled (SZ x 2)
constexpr size_t OFF_L2   = OFF_SCD  + 2*SZ;               // l2     (SZ x 2)
constexpr size_t OFF_OUT  = OFF_L2   + 2*SZ;               // out    (B x 2)
constexpr size_t OFF_ACC  = OFF_OUT  + (size_t)BATCH*2;    // accumulators (4096)
constexpr size_t OFF_W2SW = OFF_ACC  + 4096;               // 16384 bf16 (8192 fl)
constexpr size_t OFF_GSW  = OFF_W2SW + 8192;               // 16384 bf16
constexpr size_t WS_FLOATS = OFF_GSW + 8192;
// accum slots: [0..1] sq-diff sum per iter, [2..3] jac sum per iter, [4] comp
constexpr int ACC_ZNUM0 = 8;      // + i*900
constexpr int ACC_ZDEN0 = 2048;   // + i*900

// ---- small helpers ---------------------------------------------------------
DEV float softplusf(float x) { return fmaxf(x, 0.f) + log1pf(expf(-fabsf(x))); }
DEV float sigmoidf_(float x) { return 1.f / (1.f + expf(-x)); }

DEV float wave_sum(float v) {
#pragma unroll
  for (int o = 16; o > 0; o >>= 1) v += __shfl_xor(v, o, 32);
  return v;
}

DEV float block_reduce_sum(float v, float* red) {   // blockDim.x == 256
  int tid = threadIdx.x;
  __syncthreads();
  red[tid] = v; __syncthreads();
  for (int o = 128; o > 0; o >>= 1) { if (tid < o) red[tid] += red[tid+o]; __syncthreads(); }
  float r = red[0]; __syncthreads();
  return r;
}
DEV float block_reduce_max(float v, float* red) {
  int tid = threadIdx.x;
  __syncthreads();
  red[tid] = v; __syncthreads();
  for (int o = 128; o > 0; o >>= 1) { if (tid < o) red[tid] = fmaxf(red[tid], red[tid+o]); __syncthreads(); }
  float r = red[0]; __syncthreads();
  return r;
}

// ---- WMMA + fragment swizzles (ISA 7.12.2 layouts, wave32) ----------------
DEV v8f wmma_bf16(v16bf a, v16bf b, v8f c) {
  return __builtin_amdgcn_wmma_f32_16x16x32_bf16(false, a, false, b, (short)0, c,
                                                 false, false);
}

// A-matrix (16x32 bf16) fragment-swizzle: element (m,k) of a tile whose rows
// are grouped in 16-row tiles -> [tile][lane][e] with lane/e per ISA layout.
// lane = (m&15) + 16*((k>>3)&1) ; e = (k&7) + 8*((k>>4)&1)
DEV int swzA(int m, int k) {          // tiles of 16 rows, 32-wide k
  return (((m >> 4)*32 + (m & 15) + ((k & 8) ? 16 : 0)) << 4)
         + (k & 7) + ((k & 16) ? 8 : 0);
}
// B-matrix (32x16 bf16): lane = (n&15) + 16*((k>>3)&1) ; e = (k&7) + 8*((k>>4)&1)
DEV int swzB(int k, int n) {          // tiles of 16 cols, 32-deep k
  return (((n >> 4)*32 + (n & 15) + ((k & 8) ? 16 : 0)) << 4) + (k & 7) + ((k & 16) ? 8 : 0);
}
// same swizzles for a full 128x128 matrix (frag grid = (k>>5) x (n>>4))
DEV int swzB128(int k, int n) {
  return ((((k >> 5)*8 + (n >> 4))*32 + (n & 15) + ((k & 8) ? 16 : 0)) << 4)
         + (k & 7) + ((k & 16) ? 8 : 0);
}
DEV int swzA128(int m, int k) {       // 32 x 128 A operand (frag = (m>>4)*4+(k>>5))
  return ((((m >> 4)*4 + (k >> 5))*32 + (m & 15) + ((k & 8) ? 16 : 0)) << 4)
         + (k & 7) + ((k & 16) ? 8 : 0);
}
// C/D f32 16x16: reg r -> row r (+8 for lanes 16-31), col = lane&15.

// ===========================================================================
// init: zero accumulators and out, set l2 = 1/H = 0.5
// ===========================================================================
__global__ __launch_bounds__(256) void init_kernel(float* acc, float* outb, float* l2g) {
  int i = blockIdx.x*256 + threadIdx.x;
  if (i < 4096) acc[i] = 0.f;
  int j = i - 4096;
  if (j >= 0 && j < BATCH*2) outb[j] = 0.f;
  int k = i - 4096 - BATCH*2;
  if (k >= 0 && k < 2*SZ) l2g[k] = 0.5f;
}

// ===========================================================================
// raw + groupnorm: one block per batch row
// ===========================================================================
__global__ __launch_bounds__(256) void raw_xn_kernel(
    const float* __restrict__ x, const float* __restrict__ outb,
    const float* __restrict__ ln_g, const float* __restrict__ ln_b,
    float* __restrict__ raw, float* __restrict__ xn, int iter0) {
  __shared__ float nx[NX];
  __shared__ float rr[SZ];
  __shared__ float red[256];
  int row = blockIdx.x, tid = threadIdx.x;
  if (tid < NINP) nx[tid] = x[(size_t)row*NINP + tid];
  else if (tid < NX) nx[tid] = iter0 ? 0.f : outb[(size_t)row*2 + (tid - NINP)];
  __syncthreads();
  for (int p = tid; p < SZ; p += 256) {
    float v;
    if (p < 2*NTRI) {
      int q = (p < NTRI) ? p : p - NTRI;
      int a = 0, rem = q;
      while (rem >= NX - a) { rem -= NX - a; ++a; }
      int b = a + rem;
      if (p < NTRI) v = (a == b) ? nx[a] : nx[a] + nx[b];   // sum minus eye on diag
      else          v = nx[a] * nx[b];
    } else if (p < 2*NTRI + NX) v = sinf(nx[p - 2*NTRI]);
    else                        v = cosf(nx[p - 2*NTRI - NX]);
    rr[p] = v;
    raw[(size_t)row*SZ + p] = v;
  }
  __syncthreads();
  float ps = 0.f;
  for (int p = tid; p < SZ; p += 256) ps += rr[p];
  float m = block_reduce_sum(ps, red) / (float)SZ;
  float pv = 0.f;
  for (int p = tid; p < SZ; p += 256) { float d = rr[p] - m; pv += d*d; }
  float inv = rsqrtf(block_reduce_sum(pv, red) / (float)SZ + 1e-5f);
  for (int p = tid; p < SZ; p += 256)
    xn[(size_t)row*SZ + p] = (rr[p] - m) * inv * ln_g[p] + ln_b[p];
}

// ===========================================================================
// generic tiled GEMM: C[M,N] = act( A[M,K] @ W[N,K]^T + bias )
// 128x64 block tile, 8 waves x 4 fragments, bf16 WMMA, fp32 accumulate.
// Swizzled LDS tiles; float4 staging fast path for interior tiles.
// M must be a multiple of 128 (2048 or 128 here); N,K arbitrary (zero-padded).
// ===========================================================================
__global__ __launch_bounds__(256) void gemm_wt_kernel(
    const float* __restrict__ A, int lda,
    const float* __restrict__ W, int ldw,
    const float* __restrict__ bias,
    float* __restrict__ C, int ldc,
    float* __restrict__ sig,
    const float* __restrict__ dref, float* __restrict__ dacc,
    int M, int N, int K, int act, int storeC) {
  __shared__ v16bf AsV[8*32];     // 128x32 tile, fragment-swizzled
  __shared__ v16bf BsV[4*32];     // 32x64 tile, fragment-swizzled
  __bf16* As = (__bf16*)AsV;
  __bf16* Bs = (__bf16*)BsV;
  int tid = threadIdx.x;
  int m0 = blockIdx.y * 128, n0 = blockIdx.x * 64;
  int wave = tid >> 5, lane = tid & 31;
  int mq = (wave & 3) * 2;        // two m-tiles per wave
  int nh = wave >> 2;             // two n-tiles per wave
  v8f acc[2][2] = {};
  for (int kk = 0; kk < K; kk += 32) {
    __syncthreads();
    if (kk + 32 <= K && n0 + 64 <= N) {
      // fast path: float4 global loads -> packed bf16 -> b64 LDS stores
      for (int f = tid; f < 1024; f += 256) {            // A: 128 rows x 8 quads
        int m = f >> 3, k = (f & 7) * 4;
        float4 v = *(const float4*)(A + (size_t)(m0 + m)*lda + kk + k);
        v4bf w = { (__bf16)v.x, (__bf16)v.y, (__bf16)v.z, (__bf16)v.w };
        *(v4bf*)(As + swzA(m, k)) = w;
      }
      for (int f = tid; f < 512; f += 256) {             // B: 64 rows x 8 quads
        int n = f >> 3, k = (f & 7) * 4;
        float4 v = *(const float4*)(W + (size_t)(n0 + n)*ldw + kk + k);
        v4bf w = { (__bf16)v.x, (__bf16)v.y, (__bf16)v.z, (__bf16)v.w };
        *(v4bf*)(Bs + swzB(k, n)) = w;
      }
    } else {
      for (int idx = tid; idx < 128*32; idx += 256) {
        int m = idx >> 5, k = idx & 31;
        float v = (kk + k < K) ? A[(size_t)(m0 + m)*lda + kk + k] : 0.f;
        As[swzA(m, k)] = (__bf16)v;
      }
      for (int idx = tid; idx < 32*64; idx += 256) {
        int k = idx >> 6, n = idx & 63;
        float v = (n0 + n < N && kk + k < K) ? W[(size_t)(n0 + n)*ldw + kk + k] : 0.f;
        Bs[swzB(k, n)] = (__bf16)v;
      }
    }
    __syncthreads();
    v16bf a0 = AsV[(mq    )*32 + lane];
    v16bf a1 = AsV[(mq + 1)*32 + lane];
    v16bf b0 = BsV[(nh*2    )*32 + lane];
    v16bf b1 = BsV[(nh*2 + 1)*32 + lane];
    acc[0][0] = wmma_bf16(a0, b0, acc[0][0]);
    acc[0][1] = wmma_bf16(a0, b1, acc[0][1]);
    acc[1][0] = wmma_bf16(a1, b0, acc[1][0]);
    acc[1][1] = wmma_bf16(a1, b1, acc[1][1]);
  }
  int ho = (lane & 16) ? 8 : 0;
  int nl = lane & 15;
  float dsum = 0.f;
#pragma unroll
  for (int mi = 0; mi < 2; ++mi)
#pragma unroll
    for (int ni = 0; ni < 2; ++ni) {
      int n = n0 + (nh*2 + ni)*16 + nl;
      if (n < N) {
        float bv = bias ? bias[n] : 0.f;
#pragma unroll
        for (int r = 0; r < 8; ++r) {
          int m = m0 + (mq + mi)*16 + ho + r;
          float v = acc[mi][ni][r] + bv;
          if (storeC) C[(size_t)m*ldc + n] = (act == 1) ? softplusf(v) : v;
          if (sig)    sig[(size_t)m*ldc + n] = sigmoidf_(v);
          if (dacc) { float d = v - dref[(size_t)m*ldc + n]; dsum += d*d; }
        }
      }
    }
  if (dacc) {
    dsum = wave_sum(dsum);
    if (lane == 0) atomicAdd(dacc, dsum);
  }
}

// ===========================================================================
// prep: bf16-convert W2 and Gram G into global B-fragment-swizzled layout
// ===========================================================================
__global__ __launch_bounds__(256) void prep_sw_kernel(
    const float* __restrict__ W2, const float* __restrict__ G,
    __bf16* __restrict__ W2sw, __bf16* __restrict__ Gsw) {
  int i = blockIdx.x*256 + threadIdx.x;
  if (i < 16384) {
    int k = i >> 7, n = i & 127;
    W2sw[swzB128(k, n)] = (__bf16)W2[i];
  } else if (i < 32768) {
    int j = i - 16384;
    int k = j >> 7, n = j & 127;
    Gsw[swzB128(k, n)] = (__bf16)G[j];
  }
}

// ===========================================================================
// Jacobian Frobenius term:  sum_b tr(U~ G U~^T),
//   U~ = (W3 . diag(s2)) @ W2 . diag(s1)   (30x128, padded to 32)
// One wave per sample, 4 waves / block. B fragments stream from the global
// pre-swizzled bf16 W2sw / Gsw (L2-resident); U~ lives in per-wave swizzled
// LDS so second-GEMM A fragments are contiguous v16bf loads.
// ===========================================================================
__global__ __launch_bounds__(128) void kjac_kernel(
    const float* __restrict__ W3,
    const __bf16* __restrict__ W2sw, const __bf16* __restrict__ Gsw,
    const float* __restrict__ s1g, const float* __restrict__ s2g,
    float* __restrict__ jac_acc, int npass) {
  __shared__ __bf16 W3b[32*128];        // rows 30,31 zero-padded
  __shared__ v16bf  UtV[4][8*32];       // per-wave U~ in A-frag swizzle
  __shared__ float  s1s[4][128];
  __shared__ float  s2s[4][128];
  int tid = threadIdx.x, wave = tid >> 5, lane = tid & 31;
  for (int idx = tid; idx < 32*128; idx += 128) {
    int m = idx >> 7;
    W3b[idx] = (__bf16)((m < LL) ? W3[idx] : 0.f);
  }
  __syncthreads();
  __bf16* Ut = (__bf16*)UtV[wave];
  const v16bf* W2v = (const v16bf*)W2sw;
  const v16bf* Gv  = (const v16bf*)Gsw;
  int nl = lane & 15;
  int ho = (lane & 16) ? 8 : 0;
  float tacc = 0.f;
  for (int pass = 0; pass < npass; ++pass) {
    int sample = (pass*gridDim.x + blockIdx.x)*4 + wave;
    for (int k = lane; k < 128; k += 32) {
      s1s[wave][k] = s1g[(size_t)sample*128 + k];
      s2s[wave][k] = s2g[(size_t)sample*128 + k];
    }
    // A fragments of (W3 . diag(s2)) -- per-element gather + scale + cvt
    v16bf af[2][4];
#pragma unroll
    for (int mt = 0; mt < 2; ++mt)
#pragma unroll
      for (int ks = 0; ks < 4; ++ks) {
        int m = mt*16 + (lane & 15);
        int kb = ks*32 + ho;
        v16bf a;
#pragma unroll
        for (int v = 0; v < 8; ++v) {
          int k = kb + ((v & 3) << 1) + ((v & 4) << 2);
          a[2*v]   = (__bf16)((float)W3b[m*128 + k]     * s2s[wave][k]);
          a[2*v+1] = (__bf16)((float)W3b[m*128 + k + 1] * s2s[wave][k + 1]);
        }
        af[mt][ks] = a;
      }
    // GEMM1: U = A @ W2 ; scale cols by s1 ; store U~ to swizzled LDS
#pragma unroll
    for (int nc = 0; nc < 8; ++nc) {
      v8f a0 = {}, a1 = {};
#pragma unroll
      for (int ks = 0; ks < 4; ++ks) {
        v16bf bf = W2v[(ks*8 + nc)*32 + lane];
        a0 = wmma_bf16(af[0][ks], bf, a0);
        a1 = wmma_bf16(af[1][ks], bf, a1);
      }
      int n = nc*16 + nl;
      float sv = s1s[wave][n];
#pragma unroll
      for (int r = 0; r < 8; ++r) {
        Ut[swzA128(ho + r,      n)] = (__bf16)(a0[r]*sv);
        Ut[swzA128(16 + ho + r, n)] = (__bf16)(a1[r]*sv);
      }
    }
    // GEMM2: E = U~ @ G ; t += sum(E .* U~)
#pragma unroll
    for (int mt = 0; mt < 2; ++mt)
#pragma unroll
      for (int ks = 0; ks < 4; ++ks)
        af[mt][ks] = UtV[wave][(mt*4 + ks)*32 + lane];
#pragma unroll
    for (int nc = 0; nc < 8; ++nc) {
      v8f a0 = {}, a1 = {};
#pragma unroll
      for (int ks = 0; ks < 4; ++ks) {
        v16bf bf = Gv[(ks*8 + nc)*32 + lane];
        a0 = wmma_bf16(af[0][ks], bf, a0);
        a1 = wmma_bf16(af[1][ks], bf, a1);
      }
      int n = nc*16 + nl;
#pragma unroll
      for (int r = 0; r < 8; ++r) {
        tacc += a0[r] * (float)Ut[swzA128(ho + r,      n)];
        tacc += a1[r] * (float)Ut[swzA128(16 + ho + r, n)];
      }
    }
  }
  tacc = wave_sum(tacc);
  if (lane == 0) atomicAdd(jac_acc, tacc);
}

// ===========================================================================
// batch Gram stats: znum += z^T z, zden += |z|^T |z|   (30x30 each)
// ===========================================================================
__global__ __launch_bounds__(256) void zstats_kernel(
    const float* __restrict__ z, float* __restrict__ znum, float* __restrict__ zden) {
  __shared__ float zr[32];
  int tid = threadIdx.x;
  int pa[4], pb[4];
  float an[4] = {0,0,0,0}, ad[4] = {0,0,0,0};
#pragma unroll
  for (int q = 0; q < 4; ++q) {
    int p = tid + q*256;
    pa[q] = (p < 900) ? p / 30 : 0;
    pb[q] = (p < 900) ? p % 30 : 0;
  }
  for (int r = 0; r < 128; ++r) {
    int row = blockIdx.x*128 + r;
    __syncthreads();
    if (tid < LL) zr[tid] = z[(size_t)row*32 + tid];
    __syncthreads();
#pragma unroll
    for (int q = 0; q < 4; ++q) {
      int p = tid + q*256;
      if (p < 900) {
        float za = zr[pa[q]], zb = zr[pb[q]];
        an[q] += za*zb;
        ad[q] += fabsf(za)*fabsf(zb);
      }
    }
  }
#pragma unroll
  for (int q = 0; q < 4; ++q) {
    int p = tid + q*256;
    if (p < 900) { atomicAdd(&znum[p], an[q]); atomicAdd(&zden[p], ad[q]); }
  }
}

// ===========================================================================
// scalar[s][h] = sum_o ssc[h][s][o]   (one wave per (h,s))
// ===========================================================================
__global__ __launch_bounds__(256) void rowsum_kernel(
    const float* __restrict__ ssc, float* __restrict__ scal) {
  int gw = (blockIdx.x*256 + threadIdx.x) >> 5;
  int lane = threadIdx.x & 31;
  if (gw >= 2*SZ) return;
  int h = gw / SZ, s = gw % SZ;
  const float* p = ssc + ((size_t)h*SZ + s)*SZ;
  float a = 0.f;
  for (int o = lane; o < SZ; o += 32) a += p[o];
  a = wave_sum(a);
  if (lane == 0) scal[(size_t)s*2 + h] = a;
}

// ===========================================================================
// selector chain + entropy regularizer + cost accumulation (single block)
// ===========================================================================
__global__ __launch_bounds__(256) void selector_kernel(
    const float* __restrict__ qw, const float* __restrict__ qb,
    const float* __restrict__ sq,
    const float* __restrict__ sl1, const float* __restrict__ sg1, const float* __restrict__ sb1,
    const float* __restrict__ sl2, const float* __restrict__ sg2, const float* __restrict__ sb2,
    const float* __restrict__ son, const float* __restrict__ scal,
    float* __restrict__ l2g, float* __restrict__ scaled,
    float* __restrict__ acc, int iter) {
  __shared__ float gsa[900], lsa[900], selv[64], hh[128];
  __shared__ float A1[2*SZ], A2[2*SZ];
  __shared__ float red[256], misc[16];
  int tid = threadIdx.x;
  for (int p = tid; p < 900; p += 256)
    gsa[p] = acc[ACC_ZNUM0 + iter*900 + p] / fmaxf(acc[ACC_ZDEN0 + iter*900 + p], 1e-5f);
  __syncthreads();
  for (int p = tid; p < 900; p += 256) {
    int a = p / 30, b = p % 30;
    float s = qb[b];
    for (int c = 0; c < 30; ++c) s += gsa[a*30 + c] * qw[b*30 + c];
    lsa[p] = s;
  }
  __syncthreads();
  if (tid < 60) {
    int h = tid / 30, a = tid % 30;
    float s = 0.f;
    for (int j = 0; j < 30; ++j) s += lsa[a*30 + j] * sq[h*30 + j];
    selv[h*30 + a] = s;
  }
  __syncthreads();
  if (tid < 128) {
    int h = tid >> 6, o = tid & 63;
    float s = 0.f;
    for (int l = 0; l < 30; ++l) s += sl1[(h*64 + o)*30 + l] * selv[h*30 + l];
    hh[tid] = s;
  }
  __syncthreads();
  if (tid < 2) {
    float m = 0.f;
    for (int o = 0; o < 64; ++o) m += hh[tid*64 + o];
    m /= 64.f;
    float v = 0.f;
    for (int o = 0; o < 64; ++o) { float d = hh[tid*64 + o] - m; v += d*d; }
    misc[tid] = m; misc[2 + tid] = rsqrtf(v/64.f + 1e-5f);
  }
  __syncthreads();
  if (tid < 128) {
    int h = tid >> 6;
    float xg = (hh[tid] - misc[h]) * misc[2 + h];
    hh[tid] = softplusf(xg * sg1[tid] + sb1[tid]);
  }
  __syncthreads();
  for (int p = tid; p < 2*SZ; p += 256) {
    int h = p / SZ, s = p % SZ;
    float a = 0.f;
    const float* w = sl2 + (size_t)(h*SZ + s)*64;
    for (int o = 0; o < 64; ++o) a += w[o] * hh[h*64 + o];
    A1[p] = a;
  }
  __syncthreads();
  for (int h = 0; h < 2; ++h) {     // groupnorm + softmax per h over SZ
    float ps = 0.f;
    for (int s = tid; s < SZ; s += 256) ps += A1[h*SZ + s];
    float m = block_reduce_sum(ps, red) / (float)SZ;
    float pv = 0.f;
    for (int s = tid; s < SZ; s += 256) { float d = A1[h*SZ + s] - m; pv += d*d; }
    float inv = rsqrtf(block_reduce_sum(pv, red) / (float)SZ + 1e-5f);
    for (int s = tid; s < SZ; s += 256)
      A1[h*SZ + s] = (A1[h*SZ + s] - m) * inv * sg2[h*SZ + s] + sb2[h*SZ + s];
    __syncthreads();
    float pm = -1e30f;
    for (int s = tid; s < SZ; s += 256) pm = fmaxf(pm, A1[h*SZ + s]);
    float mx = block_reduce_max(pm, red);
    for (int s = tid; s < SZ; s += 256) A1[h*SZ + s] = expf(A1[h*SZ + s] - mx);
    __syncthreads();
    float pe = 0.f;
    for (int s = tid; s < SZ; s += 256) pe += A1[h*SZ + s];
    float si = 1.f / block_reduce_sum(pe, red);
    for (int s = tid; s < SZ; s += 256) A1[h*SZ + s] *= si;
    __syncthreads();
  }
  for (int h = 0; h < 2; ++h) {     // on/off gates (uses OLD l2)
    float po = 0.f;
    for (int s = tid; s < SZ; s += 256) {
      float il = l2g[s*2] * l2g[s*2 + 1];
      po += son[h*2*SZ + s]*il + son[h*2*SZ + SZ + s]*A1[h*SZ + s];
    }
    float d = block_reduce_sum(po, red);
    if (tid == 0) misc[4 + h] = sigmoidf_(d);
    __syncthreads();
  }
  for (int p = tid; p < 2*SZ; p += 256) {
    int h = p / SZ, s = p % SZ;
    float o  = misc[4 + h] * A1[p];
    float sc = o * scal[s*2 + h];
    if (fabsf(sc) <= 1e-14f) sc = 1e-14f;
    scaled[s*2 + h] = sc;
    float ln = fmaxf(o, 1e-14f);
    l2g[s*2 + h] = ln;
    A2[p] = ln;
  }
  __syncthreads();
  float g00 = 0.f, g01 = 0.f, g10 = 0.f, g11 = 0.f;
  for (int s = tid; s < SZ; s += 256) {
    float a0 = A2[s], a1 = A2[SZ + s];
    float l0 = logf(a0 + 1e-5f), l1 = logf(a1 + 1e-5f);
    g00 += a0*l0; g01 += a0*l1; g10 += a1*l0; g11 += a1*l1;
  }
  g00 = block_reduce_sum(g00, red);
  g01 = block_reduce_sum(g01, red);
  g10 = block_reduce_sum(g10, red);
  g11 = block_reduce_sum(g11, red);
  if (tid == 0) {
    float g = 0.01f * 0.5f * (-(g00 + g11)) - 0.25f * (-(g01 + g10));
    float cst = acc[iter] / ((float)BATCH * (float)SZ)
              + acc[2 + iter] / ((float)LL * (float)BATCH * (float)SZ);
    acc[4] += g + cst;
  }
}

// ===========================================================================
// out[b][h] = sum_s raw[b][s] * scaled[s][h]
// ===========================================================================
__global__ __launch_bounds__(256) void out_update_kernel(
    const float* __restrict__ raw, const float* __restrict__ scaled,
    float* __restrict__ outb) {
  __shared__ float sc[2*SZ];
  int tid = threadIdx.x;
  for (int p = tid; p < 2*SZ; p += 256) sc[p] = scaled[p];
  __syncthreads();
  int wave = tid >> 5, lane = tid & 31;
  int row = blockIdx.x*8 + wave;
  const float* rp = raw + (size_t)row*SZ;
  float a0 = 0.f, a1 = 0.f;
  for (int s = lane; s < SZ; s += 32) {
    float r = rp[s];
    a0 += r*sc[s*2];
    a1 += r*sc[s*2 + 1];
  }
  a0 = wave_sum(a0); a1 = wave_sum(a1);
  if (lane == 0) { outb[(size_t)row*2] = a0; outb[(size_t)row*2 + 1] = a1; }
}

// ===========================================================================
// finalize: d_out[0..2047] = out.sum(-1), d_out[2048] = comp
// ===========================================================================
__global__ __launch_bounds__(256) void final_kernel(
    const float* __restrict__ outb, const float* __restrict__ acc,
    float* __restrict__ dout) {
  int i = blockIdx.x*256 + threadIdx.x;
  if (i < BATCH) dout[i] = outb[(size_t)i*2] + outb[(size_t)i*2 + 1];
  if (i == BATCH) dout[BATCH] = acc[4];
}

// ===========================================================================
extern "C" void kernel_launch(void* const* d_in, const int* in_sizes, int n_in,
                              void* d_out, int out_size, void* d_ws, size_t ws_size,
                              hipStream_t stream) {
  (void)in_sizes; (void)n_in; (void)out_size;
  if (ws_size < WS_FLOATS * sizeof(float)) return;

  const float* x    = (const float*)d_in[0];
  const float* ln_g = (const float*)d_in[1];
  const float* ln_b = (const float*)d_in[2];
  const float* ew1  = (const float*)d_in[3];
  const float* eb1  = (const float*)d_in[4];
  const float* ew2  = (const float*)d_in[5];
  const float* eb2  = (const float*)d_in[6];
  const float* ew3  = (const float*)d_in[7];
  const float* eb3  = (const float*)d_in[8];
  const float* dw1  = (const float*)d_in[9];
  const float* db1  = (const float*)d_in[10];
  const float* dw2  = (const float*)d_in[11];
  const float* db2  = (const float*)d_in[12];
  const float* dw3  = (const float*)d_in[13];
  const float* db3  = (const float*)d_in[14];
  const float* qw   = (const float*)d_in[15];
  const float* qb   = (const float*)d_in[16];
  const float* sq   = (const float*)d_in[17];
  const float* sl1  = (const float*)d_in[18];
  const float* sg1  = (const float*)d_in[19];
  const float* sb1  = (const float*)d_in[20];
  const float* sl2  = (const float*)d_in[21];
  const float* sg2  = (const float*)d_in[22];
  const float* sb2  = (const float*)d_in[23];
  const float* ssc  = (const float*)d_in[24];
  const float* son  = (const float*)d_in[25];

  float* ws   = (float*)d_ws;
  float* raw  = ws + OFF_RAW;
  float* xn   = ws + OFF_XN;
  float* h1   = ws + OFF_H1;
  float* s1   = ws + OFF_S1;
  float* h2e  = ws + OFF_H2;
  float* s2   = ws + OFF_S2;
  float* z    = ws + OFF_Z;
  float* d1   = ws + OFF_D1;
  float* d2   = ws + OFF_D2;
  float* G    = ws + OFF_G;
  float* scal = ws + OFF_SCAL;
  float* scd  = ws + OFF_SCD;
  float* l2b  = ws + OFF_L2;
  float* outb = ws + OFF_OUT;
  float* acc  = ws + OFF_ACC;
  __bf16* w2sw = (__bf16*)(ws + OFF_W2SW);
  __bf16* gsw  = (__bf16*)(ws + OFF_GSW);

  init_kernel<<<(4096 + BATCH*2 + 2*SZ + 255)/256, 256, 0, stream>>>(acc, outb, l2b);

  // Gram G = W1 @ W1^T (once; W1 constant across iterations)
  gemm_wt_kernel<<<dim3(2, 1), 256, 0, stream>>>(
      ew1, SZ, ew1, SZ, nullptr, G, 128, nullptr, nullptr, nullptr,
      128, 128, SZ, 0, 1);
  // swizzled bf16 copies of W2 and G for the Jacobian kernel
  prep_sw_kernel<<<128, 256, 0, stream>>>(ew2, G, w2sw, gsw);

  for (int i = 0; i < 2; ++i) {
    raw_xn_kernel<<<BATCH, 256, 0, stream>>>(x, outb, ln_g, ln_b, raw, xn, i == 0);

    // encoder
    gemm_wt_kernel<<<dim3(2, 16), 256, 0, stream>>>(
        xn, SZ, ew1, SZ, eb1, h1, 128, s1, nullptr, nullptr,
        BATCH, 128, SZ, 1, 1);
    gemm_wt_kernel<<<dim3(2, 16), 256, 0, stream>>>(
        h1, 128, ew2, 128, eb2, h2e, 128, s2, nullptr, nullptr,
        BATCH, 128, 128, 1, 1);
    gemm_wt_kernel<<<dim3(1, 16), 256, 0, stream>>>(
        h2e, 128, ew3, 128, eb3, z, 32, nullptr, nullptr, nullptr,
        BATCH, LL, 128, 0, 1);
    // decoder; last layer fused with (xn - xhat)^2 reduction (no store)
    gemm_wt_kernel<<<dim3(2, 16), 256, 0, stream>>>(
        z, 32, dw1, LL, db1, d1, 128, nullptr, nullptr, nullptr,
        BATCH, 128, LL, 1, 1);
    gemm_wt_kernel<<<dim3(2, 16), 256, 0, stream>>>(
        d1, 128, dw2, 128, db2, d2, 128, nullptr, nullptr, nullptr,
        BATCH, 128, 128, 1, 1);
    gemm_wt_kernel<<<dim3(18, 16), 256, 0, stream>>>(
        d2, 128, dw3, 128, db3, nullptr, SZ, nullptr, xn, acc + i,
        BATCH, SZ, 128, 0, 0);

    // Jacobian Frobenius term (trace trick)
    kjac_kernel<<<256, 128, 0, stream>>>(ew3, w2sw, gsw, s1, s2, acc + 2 + i, 2);

    // batch Gram stats of z
    zstats_kernel<<<16, 256, 0, stream>>>(z, acc + ACC_ZNUM0 + i*900,
                                             acc + ACC_ZDEN0 + i*900);
    // ssc row sums
    rowsum_kernel<<<280, 256, 0, stream>>>(ssc + (size_t)i*2*SZ*SZ, scal);

    // selector chain + comp accumulation
    selector_kernel<<<1, 256, 0, stream>>>(
        qw + (size_t)i*900, qb + (size_t)i*30, sq + (size_t)i*60,
        sl1 + (size_t)i*2*64*30, sg1 + (size_t)i*128, sb1 + (size_t)i*128,
        sl2 + (size_t)i*2*SZ*64, sg2 + (size_t)i*2*SZ, sb2 + (size_t)i*2*SZ,
        son + (size_t)i*2*2*SZ, scal, l2b, scd, acc, i);

    // out = raw @ scaled
    out_update_kernel<<<BATCH/8, 256, 0, stream>>>(raw, scd, outb);
  }

  final_kernel<<<9, 256, 0, stream>>>(outb, acc, (float*)d_out);
}
